// MeshAttention_49752901156975
// MI455X (gfx1250) — matrix operand order repaired and checked by reading, not compiled, not run on hardware
//
#include <hip/hip_runtime.h>
#include <hip/hip_bf16.h>

#define BB 8
#define DD 256
#define EE 2048
#define HH 8
#define DKK 32
#define DVV 32

typedef __bf16 bf16;
typedef __attribute__((ext_vector_type(16))) __bf16 bf16x16;
typedef __attribute__((ext_vector_type(8)))  float  f32x8;
typedef __attribute__((ext_vector_type(4)))  float  f32x4;
typedef __attribute__((ext_vector_type(4)))  unsigned int u32x4;
typedef __attribute__((ext_vector_type(8)))  int    i32x8;
typedef __attribute__((ext_vector_type(4)))  int    i32x4;

union Frag { bf16x16 v; u32x4 u[2]; };

__device__ __forceinline__ bf16x16 ldfrag(const bf16* p0, const bf16* p1) {
  Frag f;
  f.u[0] = *(const u32x4*)p0;
  f.u[1] = *(const u32x4*)p1;
  return f.v;
}

__device__ __forceinline__ f32x8 wmma_bf16(bf16x16 a, bf16x16 b, f32x8 c) {
  // emits v_wmma_f32_16x16x32_bf16
  return __builtin_amdgcn_wmma_f32_16x16x32_bf16(false, a, false, b, (short)0, c, false, false);
}

__device__ __forceinline__ f32x8 zero8() {
  f32x8 z = {0.f,0.f,0.f,0.f,0.f,0.f,0.f,0.f};
  return z;
}

// ---------------------------------------------------------------------------
// TDM: 1-D tile DMA global -> LDS, 8-byte elements (bytes must be multiple of 8,
// bytes/8 <= 65535 so tile_dim0 fits in 16 bits). D# per CDNA5 ISA 8.3/8.4.
__device__ __forceinline__ void tdm_load_1d(void* lds_dst, const void* gsrc, unsigned bytes) {
  const unsigned n = bytes >> 3;                                  // 8B elements
  const unsigned long long ga = (unsigned long long)(size_t)gsrc;
  const unsigned lds = (unsigned)(size_t)lds_dst;                 // LDS address: low 32 bits of pointer
  u32x4 g0;
  g0.x = 1u;                                                      // count=1, user mode
  g0.y = lds;                                                     // lds_addr
  g0.z = (unsigned)(ga & 0xffffffffull);                          // global_addr[31:0]
  g0.w = (unsigned)((ga >> 32) & 0x01ffffffull) | (2u << 30);     // global_addr[56:32] | type=2
  i32x8 g1;
  g1[0] = (int)(3u << 16);      // workgroup_mask=0, data_size=3 (8B), no pad/iter/atomic
  g1[1] = (int)(n << 16);       // atomic_barrier_addr=0, tensor_dim0[15:0]=n
  g1[2] = (int)(1u << 16);      // tensor_dim0[31:16]=0, tensor_dim1[15:0]=1
  g1[3] = (int)(n << 16);       // tensor_dim1[31:16]=0, tile_dim0=n
  g1[4] = 0;                    // tile_dim1=0 (unused), tile_dim2=0 (1-D tile)
  g1[5] = (int)n;               // tensor_dim0_stride[31:0]
  g1[6] = 0;                    // tensor_dim0_stride[47:32]=0, tensor_dim1_stride lo=0
  g1[7] = 0;
  const i32x4 z4 = {0, 0, 0, 0};
  const i32x8 z8 = {0, 0, 0, 0, 0, 0, 0, 0};
  __builtin_amdgcn_tensor_load_to_lds(g0, g1, z4, z4, z8, 0);     // TENSORcnt++
}

// ---------------------------------------------------------------------------
// prep: weight f32 [K][N] -> bf16 transposed [N][K]
__global__ __launch_bounds__(256) void prep_w_kernel(const float* __restrict__ w,
                                                     bf16* __restrict__ wT) {
  const int n = blockIdx.x;      // 256 blocks
  const int k = threadIdx.x;     // 256 threads
  wT[n*256 + k] = (bf16)w[k*256 + n];
}

__global__ __launch_bounds__(256) void zero_ape_kernel(float* __restrict__ ape) {
  const int i = blockIdx.x*256 + threadIdx.x;
  if (i < BB*EE) ape[i] = 0.f;
}

__global__ __launch_bounds__(256) void mask_count_kernel(const unsigned char* __restrict__ mask,
                                                         float* __restrict__ mcnt) {
  const int i = blockIdx.x*256 + threadIdx.x;   // over B*E columns
  if (i >= BB*EE) return;
  const int b = i >> 11;
  const int k = i & (EE-1);
  const unsigned char* p = mask + (size_t)b*EE*EE + k;
  int s = 0;
  for (int q = 0; q < EE; ++q) s += p[(size_t)q*EE];
  mcnt[i] = (float)s;
}

// ---------------------------------------------------------------------------
// proj: transpose-gather x, LayerNorm, Q/K/V projections (bf16 WMMA)
// block = 128 threads (4 waves), each wave owns 16 e-rows. grid = B*(E/64).
__global__ __launch_bounds__(128) void proj_kernel(
    const float* __restrict__ x, const float* __restrict__ gamma, const float* __restrict__ beta,
    const bf16* __restrict__ wqT, const bf16* __restrict__ wkT, const bf16* __restrict__ wvT,
    bf16* __restrict__ qb, bf16* __restrict__ kb, bf16* __restrict__ vtb)
{
  extern __shared__ char smem[];
  bf16* sS = (bf16*)smem;                              // [4][16][256] raw s (bf16)
  bf16* sQ = (bf16*)(smem + 4*16*256*sizeof(bf16));    // [4][16][256] layernormed

  const int tid  = threadIdx.x;
  const int wave = tid >> 5, lane = tid & 31;
  const int b    = blockIdx.x >> 5;        // E/64 = 32 chunks per batch
  const int ec   = blockIdx.x & 31;
  const int ebase = ec*64 + wave*16;

  bf16* sSw = sS + wave*16*256;
  bf16* sQw = sQ + wave*16*256;

  // LayerNorm per edge-row (gather column of x, stride E)
  for (int r = 0; r < 16; ++r) {
    const int e = ebase + r;
    float v[8];
    float sum = 0.f;
    for (int j = 0; j < 8; ++j) {
      v[j] = x[(size_t)b*DD*EE + (size_t)(lane + 32*j)*EE + e];
      sum += v[j];
    }
    for (int off = 16; off > 0; off >>= 1) sum += __shfl_xor(sum, off, 32);
    const float mu = sum * (1.f/256.f);
    float vs = 0.f;
    for (int j = 0; j < 8; ++j) { const float d = v[j]-mu; vs += d*d; }
    for (int off = 16; off > 0; off >>= 1) vs += __shfl_xor(vs, off, 32);
    const float rinv = rsqrtf(vs*(1.f/256.f) + 1e-6f);
    for (int j = 0; j < 8; ++j) {
      const int d = lane + 32*j;
      sSw[r*256 + d] = (bf16)v[j];
      sQw[r*256 + d] = (bf16)((v[j]-mu)*rinv*gamma[d] + beta[d]);
    }
  }

  const int m   = lane & 15;
  const int sub = lane >> 4;          // half-wave
  const int ko  = sub ? 16 : 0;       // A-frag K base
  const int kh  = sub ? 16 : 0;       // B-frag K base

#pragma unroll
  for (int mat = 0; mat < 3; ++mat) {
    const bf16* A  = (mat == 0) ? sQw : sSw;   // only Q uses layernormed input
    const bf16* wT = (mat == 0) ? wqT : (mat == 1 ? wkT : wvT);
    const bf16* arow = A + m*256;
    for (int nt = 0; nt < 16; ++nt) {
      f32x8 acc = zero8();
      const bf16* brow = wT + (nt*16 + m)*256 + kh;
      for (int kk = 0; kk < 8; ++kk) {
        const bf16x16 a  = ldfrag(arow + kk*32 + ko, arow + kk*32 + ko + 8);
        const bf16x16 bv = ldfrag(brow + kk*32,      brow + kk*32 + 8);
        acc = wmma_bf16(a, bv, acc);
      }
      const int c  = nt*16 + m;
      const int hh = c >> 5, cc = c & 31;
      for (int i = 0; i < 8; ++i) {
        const int e = ebase + i + sub*8;
        const float val = acc[i];
        if (mat == 0)      qb [((size_t)(b*HH + hh)*EE + e)*DKK + cc] = (bf16)(val * 0.17677669529663689f); // fold 1/sqrt(dk)
        else if (mat == 1) kb [((size_t)(b*HH + hh)*EE + e)*DKK + cc] = (bf16)val;
        else               vtb[((size_t)(b*HH + hh)*DVV + cc)*EE + e] = (bf16)val;  // V transposed
      }
    }
  }
}

// ---------------------------------------------------------------------------
// attention: 1 WG per (b,h,128-query chunk); K/V^T staged into 256KB of LDS by
// the Tensor Data Mover. 8 waves, each owns one 16-row q-tile. 2-pass online
// softmax, NT attn stores.
__global__ __launch_bounds__(256) void attn_kernel(
    const bf16* __restrict__ qb, const bf16* __restrict__ kb, const bf16* __restrict__ vtb,
    const unsigned char* __restrict__ mask,
    float* __restrict__ attn_out, float* __restrict__ o_out, float* __restrict__ ape_out)
{
  extern __shared__ char smem[];
  bf16*  sK   = (bf16*)smem;                                          // [E][DK]  128KB
  bf16*  sV   = (bf16*)(smem + EE*DKK*2);                             // [DV][E]  128KB
  float* pf32 = (float*)(smem + 2*EE*DKK*2);                          // [8][16][32] 16KB
  bf16*  pb16 = (bf16*)(smem + 2*EE*DKK*2 + 8*16*32*4);               // [8][16][32]  8KB
  float* apeL = (float*)(smem + 2*EE*DKK*2 + 8*16*32*4 + 8*16*32*2);  // [E]          8KB

  const int tid = threadIdx.x;
  const int bid = blockIdx.x;
  const int qc  = bid & 15;          // E/128 = 16 chunks
  const int h   = (bid >> 4) & 7;
  const int b   = bid >> 7;

  // stage K and V^T for this (b,h) into LDS via the Tensor Data Mover
  if (tid < 32) {
    tdm_load_1d(sK, kb  + (size_t)(b*HH + h)*EE*DKK, EE*DKK*sizeof(bf16));
    tdm_load_1d(sV, vtb + (size_t)(b*HH + h)*DVV*EE, DVV*EE*sizeof(bf16));
    __builtin_amdgcn_s_wait_tensorcnt(0);
  }
  for (int i = tid; i < EE; i += 256) apeL[i] = 0.f;
  __syncthreads();

  const int wave = tid >> 5, lane = tid & 31;
  const int m   = lane & 15;
  const int sub = lane >> 4;
  const int ko  = sub ? 16 : 0;
  const int kh  = sub ? 16 : 0;
  const int q0  = qc*128 + wave*16;

  // Q A-fragment (already scaled by 1/sqrt(dk))
  const bf16* qrow = qb + ((size_t)(b*HH + h)*EE + q0 + m)*DKK;
  const bf16x16 aQ = ldfrag(qrow + ko, qrow + ko + 8);

  const unsigned char* mb = mask + (size_t)b*EE*EE;

  float rm[8], rs[8];
  for (int i = 0; i < 8; ++i) { rm[i] = -3.0e38f; rs[i] = 0.f; }

  // ---- pass 1: online row max / sum over all 128 key tiles
  for (int kt = 0; kt < EE/16; ++kt) {
    const int col = kt*16 + m;
    // prefetch mask a couple of tiles ahead (global_prefetch_b8)
    __builtin_prefetch(mb + (size_t)(q0 + sub*8)*EE + col + 64, 0, 1);
    const bf16* krow = sK + (kt*16 + m)*DKK + kh;
    const bf16x16 bK = ldfrag(krow, krow + 8);
    f32x8 S = zero8();
    S = wmma_bf16(aQ, bK, S);
    for (int i = 0; i < 8; ++i) {
      const int qr = q0 + i + sub*8;
      const float s = mb[(size_t)qr*EE + col] ? S[i] : -1e9f;
      float tm = s;
      tm = fmaxf(tm, __shfl_xor(tm, 1, 16));
      tm = fmaxf(tm, __shfl_xor(tm, 2, 16));
      tm = fmaxf(tm, __shfl_xor(tm, 4, 16));
      tm = fmaxf(tm, __shfl_xor(tm, 8, 16));
      const float nm = fmaxf(rm[i], tm);
      float ts = __expf(s - nm);
      ts += __shfl_xor(ts, 1, 16);
      ts += __shfl_xor(ts, 2, 16);
      ts += __shfl_xor(ts, 4, 16);
      ts += __shfl_xor(ts, 8, 16);
      rs[i] = rs[i]*__expf(rm[i] - nm) + ts;
      rm[i] = nm;
    }
  }

  float rinv[8];
  for (int i = 0; i < 8; ++i) rinv[i] = 1.f / rs[i];

  // ---- pass 2: recompute scores, write attn (NT), accumulate P*V + per-edge
  f32x8 acc0 = zero8(), acc1 = zero8();
  float* mypf = pf32 + wave*16*32;
  bf16*  mypb = pb16 + wave*16*32;
  float* abase = attn_out + ((size_t)(b*HH + h)*EE + q0)*EE;

  for (int kt2 = 0; kt2 < EE/32; ++kt2) {
    for (int t = 0; t < 2; ++t) {
      const int kt  = kt2*2 + t;
      const int col = kt*16 + m;
      const bf16* krow = sK + (kt*16 + m)*DKK + kh;
      const bf16x16 bK = ldfrag(krow, krow + 8);
      f32x8 S = zero8();
      S = wmma_bf16(aQ, bK, S);
      float apes = 0.f;
      for (int i = 0; i < 8; ++i) {
        const int r  = i + sub*8;
        const int qr = q0 + r;
        const unsigned char mk = mb[(size_t)qr*EE + col];
        const float s = mk ? S[i] : -1e9f;
        const float p = __expf(s - rm[i]) * rinv[i];
        if (mk) apes += p;
        mypf[r*32 + t*16 + m] = p;
        mypb[r*32 + t*16 + m] = (bf16)p;
      }
      atomicAdd(&apeL[col], apes);
    }
    // O += P @ V  (P relayout C->A via wave-private LDS)
    const bf16* prow = mypb + m*32;
    const bf16x16 aP = ldfrag(prow + ko, prow + ko + 8);
    const bf16* v0 = sV + (size_t)m*EE        + kt2*32 + kh;
    const bf16* v1 = sV + (size_t)(m + 16)*EE + kt2*32 + kh;
    acc0 = wmma_bf16(aP, ldfrag(v0, v0 + 8), acc0);
    acc1 = wmma_bf16(aP, ldfrag(v1, v1 + 8), acc1);
    // coalesced NT store of this 16x32 attn tile (lane -> row m, 16 floats)
    {
      const f32x4* src = (const f32x4*)(mypf + m*32 + sub*16);
      f32x4* dst = (f32x4*)(abase + (size_t)m*EE + kt2*32 + sub*16);
      __builtin_nontemporal_store(src[0], dst + 0);
      __builtin_nontemporal_store(src[1], dst + 1);
      __builtin_nontemporal_store(src[2], dst + 2);
      __builtin_nontemporal_store(src[3], dst + 3);
    }
  }

  // store O (f32) for the FC kernel
  float* orow = o_out + ((size_t)(b*HH + h)*EE + q0)*DVV;
  for (int i = 0; i < 8; ++i) {
    const int r = i + sub*8;
    orow[(size_t)r*DVV + m]      = acc0[i];
    orow[(size_t)r*DVV + 16 + m] = acc1[i];
  }

  __syncthreads();
  for (int i = tid; i < EE; i += 256) atomicAdd(&ape_out[(size_t)b*EE + i], apeL[i]);
}

// ---------------------------------------------------------------------------
// fc: x_out[b,d,e] = (O[b,e,:] @ w_fc)[d] + x[b,d,e]
__global__ __launch_bounds__(128) void fc_kernel(
    const float* __restrict__ o, const bf16* __restrict__ wfT,
    const float* __restrict__ x, float* __restrict__ xout)
{
  __shared__ bf16 sO[4*16*256];
  const int tid  = threadIdx.x;
  const int wave = tid >> 5, lane = tid & 31;
  const int b    = blockIdx.x >> 5;
  const int ec   = blockIdx.x & 31;
  const int ebase = ec*64 + wave*16;
  bf16* sOw = sO + wave*16*256;

  for (int idx = lane; idx < 16*256; idx += 32) {
    const int r = idx >> 8, c = idx & 255;
    const int hh = c >> 5, dv = c & 31;
    sOw[idx] = (bf16)o[((size_t)(b*HH + hh)*EE + ebase + r)*DVV + dv];
  }

  const int m   = lane & 15;
  const int sub = lane >> 4;
  const int ko  = sub ? 16 : 0;
  const int kh  = sub ? 16 : 0;
  const bf16* arow = sOw + m*256;

  for (int nt = 0; nt < 16; ++nt) {
    f32x8 acc = zero8();
    const bf16* brow = wfT + (nt*16 + m)*256 + kh;
    for (int kk = 0; kk < 8; ++kk) {
      const bf16x16 a  = ldfrag(arow + kk*32 + ko, arow + kk*32 + ko + 8);
      const bf16x16 bv = ldfrag(brow + kk*32,      brow + kk*32 + 8);
      acc = wmma_bf16(a, bv, acc);
    }
    const int d = nt*16 + m;
    for (int i = 0; i < 8; ++i) {
      const int e = ebase + i + sub*8;
      const size_t xi = (size_t)b*DD*EE + (size_t)d*EE + e;
      xout[xi] = acc[i] + x[xi];
    }
  }
}

__global__ __launch_bounds__(256) void ape_finalize_kernel(float* __restrict__ ape,
                                                           const float* __restrict__ mcnt) {
  const int i = blockIdx.x*256 + threadIdx.x;
  if (i < BB*EE) ape[i] = ape[i] / (8.0f * mcnt[i]);
}

// ---------------------------------------------------------------------------
extern "C" void kernel_launch(void* const* d_in, const int* in_sizes, int n_in,
                              void* d_out, int out_size, void* d_ws, size_t ws_size,
                              hipStream_t stream) {
  const float*         x     = (const float*)d_in[0];
  const unsigned char* mask  = (const unsigned char*)d_in[1];  // jnp bool -> 1 byte
  const float*         w_qs  = (const float*)d_in[2];
  const float*         w_ks  = (const float*)d_in[3];
  const float*         w_vs  = (const float*)d_in[4];
  const float*         w_fc  = (const float*)d_in[5];
  const float*         gamma = (const float*)d_in[6];
  const float*         beta  = (const float*)d_in[7];

  char* ws = (char*)d_ws;
  const size_t WSZ  = 256*256*sizeof(bf16);              // 128KB per weight
  const size_t QKV  = (size_t)BB*HH*EE*DKK*sizeof(bf16); // 8.38MB each
  bf16*  wqT  = (bf16*)(ws);
  bf16*  wkT  = (bf16*)(ws + WSZ);
  bf16*  wvT  = (bf16*)(ws + 2*WSZ);
  bf16*  wfT  = (bf16*)(ws + 3*WSZ);
  bf16*  qb   = (bf16*)(ws + 4*WSZ);
  bf16*  kb   = (bf16*)(ws + 4*WSZ + QKV);
  bf16*  vtb  = (bf16*)(ws + 4*WSZ + 2*QKV);
  float* o    = (float*)(ws + 4*WSZ + 3*QKV);
  float* mcnt = (float*)(ws + 4*WSZ + 3*QKV + (size_t)BB*HH*EE*DVV*sizeof(float));

  float* xout = (float*)d_out;
  float* attn = xout + (size_t)BB*DD*EE;
  float* ape  = attn + (size_t)BB*HH*EE*EE;

  prep_w_kernel<<<256, 256, 0, stream>>>(w_qs, wqT);
  prep_w_kernel<<<256, 256, 0, stream>>>(w_ks, wkT);
  prep_w_kernel<<<256, 256, 0, stream>>>(w_vs, wvT);
  prep_w_kernel<<<256, 256, 0, stream>>>(w_fc, wfT);
  zero_ape_kernel<<<(BB*EE + 255)/256, 256, 0, stream>>>(ape);
  mask_count_kernel<<<(BB*EE + 255)/256, 256, 0, stream>>>(mask, mcnt);

  const size_t proj_smem = 2 * 4*16*256*sizeof(bf16);                       // 64KB
  proj_kernel<<<BB*(EE/64), 128, proj_smem, stream>>>(x, gamma, beta, wqT, wkT, wvT, qb, kb, vtb);

  const size_t attn_smem = 2*EE*DKK*sizeof(bf16)      // K + V^T   256KB
                         + 8*16*32*sizeof(float)      // P f32      16KB
                         + 8*16*32*sizeof(bf16)       // P bf16      8KB
                         + EE*sizeof(float);          // ape accum   8KB  => 288KB (<320KB WGP LDS)
  attn_kernel<<<BB*HH*(EE/128), 256, attn_smem, stream>>>(qb, kb, vtb, mask, attn, o, ape);

  fc_kernel<<<BB*(EE/64), 128, 0, stream>>>(o, wfT, x, xout);
  ape_finalize_kernel<<<(BB*EE + 255)/256, 256, 0, stream>>>(ape, mcnt);
}
